// SimplifiedDATBlock_11012296147316
// MI455X (gfx1250) — compile-verified, matrix operand order checked by reading, not run on hardware
//
#include <hip/hip_runtime.h>
#include <hip/hip_bf16.h>
#include <math.h>

typedef __attribute__((ext_vector_type(16))) _Float16 v16h;
typedef __attribute__((ext_vector_type(8)))  _Float16 v8h;
typedef __attribute__((ext_vector_type(8)))  float    v8f;

#define Bc    8
#define Himg  128
#define Wimg  128
#define Limg  16384      // 128*128
#define Cc    192
#define NHc   6
#define HDh   32         // head dim
#define HDIMc 384
#define REDc  24
#define Mrows 131072     // B * L

__device__ __forceinline__ float gelu_f(float x) {
    return 0.5f * x * (1.0f + erff(x * 0.70710678118654752f));
}

union V16 { v16h v; v8h h[2]; };

__device__ __forceinline__ v8h cvt8(float4 a, float4 b) {
    v8h r;
    r[0] = (_Float16)a.x; r[1] = (_Float16)a.y; r[2] = (_Float16)a.z; r[3] = (_Float16)a.w;
    r[4] = (_Float16)b.x; r[5] = (_Float16)b.y; r[6] = (_Float16)b.z; r[7] = (_Float16)b.w;
    return r;
}

// ---------------------------------------------------------------------------
// LayerNorm: one wave per row, 8 rows per 256-thread block
// ---------------------------------------------------------------------------
__global__ void ln_kernel(const float* __restrict__ x, const float* __restrict__ w,
                          const float* __restrict__ b, float* __restrict__ y,
                          int rows, int Cx) {
    int wave = threadIdx.x >> 5;
    int lane = threadIdx.x & 31;
    int row  = blockIdx.x * 8 + wave;
    if (row >= rows) return;
    const float* xr = x + (size_t)row * Cx;
    float s = 0.f;
    for (int j = lane; j < Cx; j += 32) s += xr[j];
    for (int o = 16; o > 0; o >>= 1) s += __shfl_xor(s, o, 32);
    float mean = s / (float)Cx;
    float v = 0.f;
    for (int j = lane; j < Cx; j += 32) { float d = xr[j] - mean; v += d * d; }
    for (int o = 16; o > 0; o >>= 1) v += __shfl_xor(v, o, 32);
    float inv = rsqrtf(v / (float)Cx + 1e-5f);
    float* yr = y + (size_t)row * Cx;
    for (int j = lane; j < Cx; j += 32) yr[j] = (xr[j] - mean) * inv * w[j] + b[j];
}

// ---------------------------------------------------------------------------
// WMMA GEMM, 64x64 block tile, 128 threads (4 waves), wave tile 32x32 (2x2 WMMA).
// A tile row-major, B tile transposed in LDS -> every fragment = 2x ds_load_b128.
// Requires M%64==0, N%64==0, K%32==0 (true for all uses here).
// ---------------------------------------------------------------------------
template<int ACT, bool BIAS, bool RESID>
__global__ __launch_bounds__(128)
void gemm64_kernel(const float* __restrict__ A, const float* __restrict__ Bm,
                   const float* __restrict__ bias, const float* __restrict__ resid,
                   float* __restrict__ Cout, int M, int N, int K) {
    __shared__ __align__(16) _Float16 As[64][40];   // [m][k], stride 80B (16B aligned)
    __shared__ __align__(16) _Float16 Bt[64][40];   // [n][k] (B transposed)
    const int t    = threadIdx.x;
    const int lane = t & 31;
    const int wv   = t >> 5;           // 0..3
    const int wm   = (wv >> 1) * 32;   // wave row offset in 64x64 tile
    const int wn   = (wv & 1) * 32;    // wave col offset
    const int m0   = blockIdx.y * 64;
    const int n0   = blockIdx.x * 64;
    const int lm   = lane & 15;
    const int kh   = lane >> 4;

    v8f acc[2][2];
    #pragma unroll
    for (int i = 0; i < 2; ++i)
        #pragma unroll
        for (int j = 0; j < 2; ++j)
            acc[i][j] = (v8f){0.f,0.f,0.f,0.f,0.f,0.f,0.f,0.f};

    for (int k0 = 0; k0 < K; k0 += 32) {
        // stage A: 64 rows x 32 k = 256 (row, 8k-quarter) units, 2 per thread
        #pragma unroll
        for (int uu = 0; uu < 2; ++uu) {
            int u = t + uu * 128;
            int r = u >> 2, q = (u & 3) * 8;
            const float4* src = (const float4*)&A[(size_t)(m0 + r) * K + k0 + q];
            *(v8h*)&As[r][q] = cvt8(src[0], src[1]);
        }
        // stage B transposed: 32 k-rows x 64 n = 256 (k, 8n-group) units, 2 per thread
        #pragma unroll
        for (int uu = 0; uu < 2; ++uu) {
            int u = t + uu * 128;
            int r = u >> 3, nb = (u & 7) * 8;
            const float4* src = (const float4*)&Bm[(size_t)(k0 + r) * N + n0 + nb];
            float4 f0 = src[0], f1 = src[1];
            Bt[nb + 0][r] = (_Float16)f0.x; Bt[nb + 1][r] = (_Float16)f0.y;
            Bt[nb + 2][r] = (_Float16)f0.z; Bt[nb + 3][r] = (_Float16)f0.w;
            Bt[nb + 4][r] = (_Float16)f1.x; Bt[nb + 5][r] = (_Float16)f1.y;
            Bt[nb + 6][r] = (_Float16)f1.z; Bt[nb + 7][r] = (_Float16)f1.w;
        }
        if (k0 + 32 < K) {
            __builtin_prefetch(&A[(size_t)(m0 + (t >> 1)) * K + k0 + 32 + (t & 1) * 16], 0, 1);
            __builtin_prefetch(&Bm[(size_t)(k0 + 32 + (t & 31)) * N + n0 + (t >> 5) * 16], 0, 1);
        }
        __syncthreads();

        V16 af[2], bf[2];
        #pragma unroll
        for (int i = 0; i < 2; ++i) {
            int row = wm + i * 16 + lm;
            af[i].h[0] = *(const v8h*)&As[row][kh * 8];
            af[i].h[1] = *(const v8h*)&As[row][16 + kh * 8];
            int col = wn + i * 16 + lm;
            bf[i].h[0] = *(const v8h*)&Bt[col][kh * 8];
            bf[i].h[1] = *(const v8h*)&Bt[col][16 + kh * 8];
        }
        #pragma unroll
        for (int i = 0; i < 2; ++i)
            #pragma unroll
            for (int j = 0; j < 2; ++j)
                acc[i][j] = __builtin_amdgcn_wmma_f32_16x16x32_f16(
                    false, af[i].v, false, bf[j].v, (short)0, acc[i][j], false, false);
        __syncthreads();
    }

    #pragma unroll
    for (int i = 0; i < 2; ++i)
        #pragma unroll
        for (int j = 0; j < 2; ++j)
            #pragma unroll
            for (int e = 0; e < 8; ++e) {
                int r = m0 + wm + i * 16 + e + 8 * kh;  // C/D: VGPR e -> M=e / M=8+e
                int c = n0 + wn + j * 16 + lm;
                float v = acc[i][j][e];
                if (BIAS)  v += bias[c];
                if (ACT)   v  = gelu_f(v);
                if (RESID) v += resid[(size_t)r * N + c];
                Cout[(size_t)r * N + c] = v;
            }
}

// ---------------------------------------------------------------------------
// Window attention: one block per (window, head), 128 threads = 4 waves.
// q/k row-major f16 in LDS, v transposed -> all WMMA fragments are b128 loads.
// ---------------------------------------------------------------------------
__global__ __launch_bounds__(128)
void win_attn_kernel(const float* __restrict__ qkv, const float* __restrict__ abias,
                     float* __restrict__ out) {
    const int h   = blockIdx.x % NHc;
    const int win = blockIdx.x / NHc;
    const int b   = win >> 8;
    const int wy  = (win >> 4) & 15;
    const int wx  = win & 15;
    const int t    = threadIdx.x;
    const int lane = t & 31;
    const int wv   = t >> 5;
    const int lm   = lane & 15;
    const int kh   = lane >> 4;

    __shared__ __align__(16) _Float16 qs[64][40];    // [token][d]
    __shared__ __align__(16) _Float16 ks[64][40];    // [token][d]
    __shared__ __align__(16) _Float16 vst[32][72];   // [d][token]  (v transposed)
    __shared__ __align__(16) float    lg[64][68];
    __shared__ __align__(16) _Float16 aw[64][72];
    __shared__ float pmax[64][2];
    __shared__ float psum[64][2];

    const float scale = 0.17677669529663687f;   // 1/sqrt(32)

    // stage q,k (vector stores) and v (transposed, scalar stores)
    #pragma unroll
    for (int uu = 0; uu < 2; ++uu) {
        int u = t + uu * 128;                  // 256 units: (token, 8d-quarter)
        int n = u >> 2, q = (u & 3) * 8;
        int yy = wy * 8 + (n >> 3), xx = wx * 8 + (n & 7);
        size_t tok = (size_t)b * Limg + (size_t)yy * Wimg + xx;
        const float* base = qkv + tok * (3 * Cc) + h * HDh + q;
        float4 q0 = *(const float4*)(base);
        float4 q1 = *(const float4*)(base + 4);
        q0.x *= scale; q0.y *= scale; q0.z *= scale; q0.w *= scale;
        q1.x *= scale; q1.y *= scale; q1.z *= scale; q1.w *= scale;
        *(v8h*)&qs[n][q] = cvt8(q0, q1);
        float4 k0 = *(const float4*)(base + Cc);
        float4 k1 = *(const float4*)(base + Cc + 4);
        *(v8h*)&ks[n][q] = cvt8(k0, k1);
        float4 v0 = *(const float4*)(base + 2 * Cc);
        float4 v1 = *(const float4*)(base + 2 * Cc + 4);
        vst[q + 0][n] = (_Float16)v0.x; vst[q + 1][n] = (_Float16)v0.y;
        vst[q + 2][n] = (_Float16)v0.z; vst[q + 3][n] = (_Float16)v0.w;
        vst[q + 4][n] = (_Float16)v1.x; vst[q + 5][n] = (_Float16)v1.y;
        vst[q + 6][n] = (_Float16)v1.z; vst[q + 7][n] = (_Float16)v1.w;
    }
    __syncthreads();

    // logits[64,64] = (q*scale) @ k^T + bias[h]
    {
        V16 af;
        af.h[0] = *(const v8h*)&qs[wv * 16 + lm][kh * 8];
        af.h[1] = *(const v8h*)&qs[wv * 16 + lm][16 + kh * 8];
        #pragma unroll
        for (int nt = 0; nt < 4; ++nt) {
            V16 bf;
            bf.h[0] = *(const v8h*)&ks[nt * 16 + lm][kh * 8];
            bf.h[1] = *(const v8h*)&ks[nt * 16 + lm][16 + kh * 8];
            v8f c = (v8f){0.f,0.f,0.f,0.f,0.f,0.f,0.f,0.f};
            c = __builtin_amdgcn_wmma_f32_16x16x32_f16(false, af.v, false, bf.v,
                                                       (short)0, c, false, false);
            #pragma unroll
            for (int i = 0; i < 8; ++i) {
                int m = wv * 16 + i + 8 * kh;
                int n = nt * 16 + lm;
                lg[m][n] = c[i] + abias[(size_t)h * 4096 + (size_t)m * 64 + n];
            }
        }
    }
    __syncthreads();

    // softmax: 2 threads per row, float4-vectorized passes over LDS
    {
        int row = t >> 1, half = t & 1;
        float* lr = &lg[row][half * 32];
        float mx = -1e30f;
        #pragma unroll
        for (int j = 0; j < 8; ++j) {
            float4 v4 = *(const float4*)&lr[j * 4];
            mx = fmaxf(mx, fmaxf(fmaxf(v4.x, v4.y), fmaxf(v4.z, v4.w)));
        }
        pmax[row][half] = mx;
        __syncthreads();
        float m = fmaxf(pmax[row][0], pmax[row][1]);
        float s = 0.f;
        #pragma unroll
        for (int j = 0; j < 8; ++j) {
            float4 v4 = *(const float4*)&lr[j * 4];
            v4.x = __expf(v4.x - m); v4.y = __expf(v4.y - m);
            v4.z = __expf(v4.z - m); v4.w = __expf(v4.w - m);
            *(float4*)&lr[j * 4] = v4;
            s += v4.x + v4.y + v4.z + v4.w;
        }
        psum[row][half] = s;
        __syncthreads();
        float inv = 1.f / (psum[row][0] + psum[row][1]);
        #pragma unroll
        for (int j = 0; j < 4; ++j) {
            float4 a = *(const float4*)&lr[j * 8];
            float4 bq = *(const float4*)&lr[j * 8 + 4];
            a.x *= inv; a.y *= inv; a.z *= inv; a.w *= inv;
            bq.x *= inv; bq.y *= inv; bq.z *= inv; bq.w *= inv;
            *(v8h*)&aw[row][half * 32 + j * 8] = cvt8(a, bq);
        }
    }
    __syncthreads();

    // out[64,32] = attn @ v   (K=64 in two steps; B fragment contiguous via vst)
    #pragma unroll
    for (int nt = 0; nt < 2; ++nt) {
        v8f c = (v8f){0.f,0.f,0.f,0.f,0.f,0.f,0.f,0.f};
        #pragma unroll
        for (int ki = 0; ki < 2; ++ki) {
            int kk = ki * 32;
            V16 af, bf;
            af.h[0] = *(const v8h*)&aw[wv * 16 + lm][kk + kh * 8];
            af.h[1] = *(const v8h*)&aw[wv * 16 + lm][kk + 16 + kh * 8];
            bf.h[0] = *(const v8h*)&vst[nt * 16 + lm][kk + kh * 8];
            bf.h[1] = *(const v8h*)&vst[nt * 16 + lm][kk + 16 + kh * 8];
            c = __builtin_amdgcn_wmma_f32_16x16x32_f16(false, af.v, false, bf.v,
                                                       (short)0, c, false, false);
        }
        #pragma unroll
        for (int i = 0; i < 8; ++i) {
            int m = wv * 16 + i + 8 * kh;
            int yy = wy * 8 + (m >> 3), xx = wx * 8 + (m & 7);
            size_t tok = (size_t)b * Limg + (size_t)yy * Wimg + xx;
            out[tok * Cc + h * HDh + nt * 16 + lm] = c[i];
        }
    }
}

// ---------------------------------------------------------------------------
// Depthwise 3x3 'SAME' conv, channel-last, float4 over channels
// ---------------------------------------------------------------------------
template<bool GELU>
__global__ void dwconv4_kernel(const float* __restrict__ x, const float* __restrict__ w,
                               float* __restrict__ y, int Cx) {
    const int C4 = Cx >> 2;
    size_t idx = (size_t)blockIdx.x * blockDim.x + threadIdx.x;
    size_t total = (size_t)Bc * Limg * C4;
    if (idx >= total) return;
    int c4 = (int)(idx % C4);
    size_t bl = idx / C4;
    int l = (int)(bl % Limg);
    int b = (int)(bl / Limg);
    int y0 = l >> 7, x0 = l & 127;
    int c = c4 * 4;
    float4 s = make_float4(0.f, 0.f, 0.f, 0.f);
    #pragma unroll
    for (int dy = -1; dy <= 1; ++dy) {
        int yy = y0 + dy;
        if (yy < 0 || yy > 127) continue;
        #pragma unroll
        for (int dx = -1; dx <= 1; ++dx) {
            int xx = x0 + dx;
            if (xx < 0 || xx > 127) continue;
            int tap = (dy + 1) * 3 + (dx + 1);
            float4 xv = ((const float4*)x)[(size_t)((size_t)b * Limg + (size_t)yy * Wimg + xx) * C4 + c4];
            s.x += xv.x * w[(c + 0) * 9 + tap];
            s.y += xv.y * w[(c + 1) * 9 + tap];
            s.z += xv.z * w[(c + 2) * 9 + tap];
            s.w += xv.w * w[(c + 3) * 9 + tap];
        }
    }
    if (GELU) { s.x = gelu_f(s.x); s.y = gelu_f(s.y); s.z = gelu_f(s.z); s.w = gelu_f(s.w); }
    ((float4*)y)[idx] = s;
}

// pooled[b,c] = mean over L of conv_feat[b,l,c]
__global__ void pool_kernel(const float* __restrict__ cf, float* __restrict__ pooled) {
    int bc = blockIdx.x;
    int b = bc / Cc, c = bc % Cc;
    const float* p = cf + (size_t)b * Limg * Cc + c;
    float s = 0.f;
    for (int l = threadIdx.x; l < Limg; l += 256) s += p[(size_t)l * Cc];
    __shared__ float red[256];
    red[threadIdx.x] = s;
    __syncthreads();
    for (int o = 128; o > 0; o >>= 1) {
        if (threadIdx.x < o) red[threadIdx.x] += red[threadIdx.x + o];
        __syncthreads();
    }
    if (threadIdx.x == 0) pooled[bc] = red[0] / (float)Limg;
}

// cm[b,c] = sigmoid(gelu(pooled @ w1^T) @ w2^T)
__global__ void gate_kernel(const float* __restrict__ pooled, const float* __restrict__ w1,
                            const float* __restrict__ w2, float* __restrict__ cm) {
    int b = blockIdx.x, t = threadIdx.x;   // 192 threads
    __shared__ float pl[Cc];
    __shared__ float hh[REDc];
    pl[t] = pooled[b * Cc + t];
    __syncthreads();
    if (t < REDc) {
        float s = 0.f;
        for (int c = 0; c < Cc; ++c) s += pl[c] * w1[t * Cc + c];
        hh[t] = gelu_f(s);
    }
    __syncthreads();
    float s = 0.f;
    for (int r = 0; r < REDc; ++r) s += hh[r] * w2[t * REDc + r];
    cm[b * Cc + t] = 1.f / (1.f + __expf(-s));
}

// x2 = x + attn_feat * cm[b,:] + conv_feat  (float4)
__global__ void combine4_kernel(const float* __restrict__ x, const float* __restrict__ af,
                                const float* __restrict__ cf, const float* __restrict__ cm,
                                float* __restrict__ out) {
    const int C4 = Cc >> 2;
    size_t idx = (size_t)blockIdx.x * blockDim.x + threadIdx.x;
    size_t total = (size_t)Bc * Limg * C4;
    if (idx >= total) return;
    int c4 = (int)(idx % C4);
    int b = (int)(idx / ((size_t)Limg * C4));
    float4 xv = ((const float4*)x)[idx];
    float4 av = ((const float4*)af)[idx];
    float4 cv = ((const float4*)cf)[idx];
    float4 gm = *(const float4*)&cm[b * Cc + c4 * 4];
    float4 r;
    r.x = xv.x + av.x * gm.x + cv.x;
    r.y = xv.y + av.y * gm.y + cv.y;
    r.z = xv.z + av.z * gm.z + cv.z;
    r.w = xv.w + av.w * gm.w + cv.w;
    ((float4*)out)[idx] = r;
}

// ---------------------------------------------------------------------------
extern "C" void kernel_launch(void* const* d_in, const int* in_sizes, int n_in,
                              void* d_out, int out_size, void* d_ws, size_t ws_size,
                              hipStream_t stream) {
    (void)in_sizes; (void)n_in; (void)out_size; (void)ws_size;
    const float* x         = (const float*)d_in[0];
    const float* n1_w      = (const float*)d_in[3];
    const float* n1_b      = (const float*)d_in[4];
    const float* qkv_w     = (const float*)d_in[5];
    const float* proj_w    = (const float*)d_in[6];
    const float* proj_b    = (const float*)d_in[7];
    const float* attn_bias = (const float*)d_in[8];
    const float* conv_w    = (const float*)d_in[9];
    const float* cg_w1     = (const float*)d_in[10];
    const float* cg_w2     = (const float*)d_in[11];
    const float* n2_w      = (const float*)d_in[12];
    const float* n2_b      = (const float*)d_in[13];
    const float* fc1_w     = (const float*)d_in[14];
    const float* smix_w    = (const float*)d_in[15];
    const float* fc2_w     = (const float*)d_in[16];
    float* out = (float*)d_out;
    float* ws  = (float*)d_ws;

    const size_t LC = (size_t)Mrows * Cc;     // 25,165,824 floats
    float* N1     = ws;                       // [0, 1LC)  ; reused for N2
    float* QKV    = ws + LC;                  // [1LC, 4LC); reused for FFN hidden (2LC)
    float* CONV   = ws + 4 * LC;              // [4LC, 5LC)
    float* ATTNO  = ws + 5 * LC;              // [5LC, 6LC)
    float* ATTNF  = ws + 6 * LC;              // [6LC, 7LC)
    float* X2     = ws + 7 * LC;              // [7LC, 8LC)
    float* POOLED = ws + 8 * LC;              // small
    float* CM     = POOLED + 2048;
    float* Hbuf   = QKV;                      // [1LC, 3LC) = [Mrows, 384]
    float* H2     = CONV;                     // [4LC, 6LC) = [Mrows, 384]

    // 1. n1 = LN(x)
    ln_kernel<<<Mrows / 8, 256, 0, stream>>>(x, n1_w, n1_b, N1, Mrows, Cc);
    // 2. qkv = n1 @ qkv_w                    [Mrows, 576]
    gemm64_kernel<0, false, false><<<dim3(576 / 64, Mrows / 64), 128, 0, stream>>>(
        N1, qkv_w, nullptr, nullptr, QKV, Mrows, 3 * Cc, Cc);
    // 3. conv_feat = gelu(dwconv(n1))
    {
        size_t tot = LC / 4;
        dwconv4_kernel<true><<<(unsigned)((tot + 255) / 256), 256, 0, stream>>>(N1, conv_w, CONV, Cc);
    }
    // 4. window attention heads -> ATTNO     [Mrows, 192]
    win_attn_kernel<<<2048 * NHc, 128, 0, stream>>>(QKV, attn_bias, ATTNO);
    // 5. attn_feat = ATTNO @ proj_w + proj_b
    gemm64_kernel<0, true, false><<<dim3(Cc / 64, Mrows / 64), 128, 0, stream>>>(
        ATTNO, proj_w, proj_b, nullptr, ATTNF, Mrows, Cc, Cc);
    // 6. pooled = mean_L(conv_feat)
    pool_kernel<<<Bc * Cc, 256, 0, stream>>>(CONV, POOLED);
    // 7. cm = sigmoid(gelu(pooled @ w1^T) @ w2^T)
    gate_kernel<<<Bc, Cc, 0, stream>>>(POOLED, cg_w1, cg_w2, CM);
    // 8. x2 = x + attn_feat*cm + conv_feat
    combine4_kernel<<<(unsigned)((LC / 4 + 255) / 256), 256, 0, stream>>>(x, ATTNF, CONV, CM, X2);
    // 9. n2 = LN(x2)   (reuse N1 buffer)
    ln_kernel<<<Mrows / 8, 256, 0, stream>>>(X2, n2_w, n2_b, N1, Mrows, Cc);
    // 10. h = gelu(n2 @ fc1_w)               [Mrows, 384]
    gemm64_kernel<1, false, false><<<dim3(HDIMc / 64, Mrows / 64), 128, 0, stream>>>(
        N1, fc1_w, nullptr, nullptr, Hbuf, Mrows, HDIMc, Cc);
    // 11. h2 = dwconv(h, smix_w)             [Mrows, 384]
    {
        size_t tot = (size_t)Mrows * HDIMc / 4;
        dwconv4_kernel<false><<<(unsigned)((tot + 255) / 256), 256, 0, stream>>>(Hbuf, smix_w, H2, HDIMc);
    }
    // 12. out = x2 + h2 @ fc2_w
    gemm64_kernel<0, false, true><<<dim3(Cc / 64, Mrows / 64), 128, 0, stream>>>(
        H2, fc2_w, nullptr, X2, out, Mrows, Cc, HDIMc);
}
